// GNNSat_V2_2_18940805776102
// MI455X (gfx1250) — matrix operand order repaired and checked by reading, not compiled, not run on hardware
//
#include <hip/hip_runtime.h>
#include <hip/hip_bf16.h>
#include <math.h>

// ---------------------------------------------------------------------------
// CDNA5 (gfx1250) implementation of a 2-layer GAT network.
// WMMA (v_wmma_f32_16x16x32_f16) for the dense [N,64]x[64,C] GEMMs.
// Edge passes are memory-lean: scalar logit per edge, L2-resident atomics.
// ---------------------------------------------------------------------------

typedef __attribute__((ext_vector_type(16))) _Float16 v16h;
typedef __attribute__((ext_vector_type(8)))  float    v8f;

#define NEG_GAT 0.2f
#define NEG_ACT 0.01f
#define BN_EPS  1e-5f

__device__ __forceinline__ float leakyf(float v, float s) {
    return v >= 0.f ? v : v * s;
}

// float atomic max via int punning (handles mixed signs, init = -inf)
__device__ __forceinline__ void atomicMaxFloat(float* addr, float val) {
    if (val >= 0.f) atomicMax((int*)addr, __float_as_int(val));
    else            atomicMin((unsigned int*)addr, __float_as_uint(val));
}

// ---- WMMA fragment loaders ------------------------------------------------
// A: 16x32 f16, M x K.  Lanes 0-15 -> M=lane, K base 0; lanes 16-31 -> K base 8.
// VGPR v<4 holds K = base+2v,2v+1 ; v>=4 holds K = base+16+2(v-4),+1.
// Per lane this is two contiguous 8-float runs -> 4x b128 loads.
__device__ __forceinline__ v16h load_a_frag(const float* __restrict__ src,
                                            int row0, int k0) {
    int l = threadIdx.x & 31;
    int m = l & 15;
    int base = (l >= 16) ? 8 : 0;
    const float4* p =
        (const float4*)(src + (size_t)(row0 + m) * 64 + k0 + base);
    float4 g0 = p[0];            // K = base+0..3
    float4 g1 = p[1];            // K = base+4..7
    float4 g2 = p[4];            // K = base+16..19
    float4 g3 = p[5];            // K = base+20..23
    v16h a;
    a[0]  = (_Float16)g0.x; a[1]  = (_Float16)g0.y;
    a[2]  = (_Float16)g0.z; a[3]  = (_Float16)g0.w;
    a[4]  = (_Float16)g1.x; a[5]  = (_Float16)g1.y;
    a[6]  = (_Float16)g1.z; a[7]  = (_Float16)g1.w;
    a[8]  = (_Float16)g2.x; a[9]  = (_Float16)g2.y;
    a[10] = (_Float16)g2.z; a[11] = (_Float16)g2.w;
    a[12] = (_Float16)g3.x; a[13] = (_Float16)g3.y;
    a[14] = (_Float16)g3.z; a[15] = (_Float16)g3.w;
    return a;
}

// B: 32x16 f16, K x N (W row-major [K][C]). Lanes 0-15 -> N=lane, K base 0;
// lanes 16-31 -> N=lane-16, K base 8. Elem s<8: K=base+s; s>=8: K=base+16+(s-8).
template <int C>
__device__ __forceinline__ v16h load_b_frag(const float* __restrict__ W,
                                            int col0, int k0) {
    int l = threadIdx.x & 31;
    int n = col0 + (l & 15);
    int kb = (l >= 16) ? 8 : 0;
    const float* p = W + (size_t)(k0 + kb) * C + n;
    v16h b;
#pragma unroll
    for (int v = 0; v < 8; ++v) {
        b[2 * v]     = (_Float16)p[(2 * v + (v >= 4 ? 8 : 0)) * C];
        b[2 * v + 1] = (_Float16)p[(2 * v + 1 + (v >= 4 ? 8 : 0)) * C];
    }
    return b;
}

// ---- WMMA GEMM: out[N,C] = act(in[N,64] @ W[64,C] + bias) -----------------
// One wave per (16-row, 16-col) tile. K=64 -> 2 WMMAs, f32 accumulate.
// C templated so store addressing folds to immediate offsets.
template <int C>
__global__ void gemm64_wmma(const float* __restrict__ in,
                            const float* __restrict__ W,
                            const float* __restrict__ bias,
                            float* __restrict__ out,
                            int N, float slope /* <0 => no act */) {
    int row0 = blockIdx.x * 16;
    int col0 = blockIdx.y * 16;

    v16h a0 = load_a_frag(in, row0, 0);
    v16h a1 = load_a_frag(in, row0, 32);
    v16h b0 = load_b_frag<C>(W, col0, 0);
    v16h b1 = load_b_frag<C>(W, col0, 32);

    v8f c = {};
    c = __builtin_amdgcn_wmma_f32_16x16x32_f16(false, a0, false, b0,
                                               (short)0, c, false, false);
    c = __builtin_amdgcn_wmma_f32_16x16x32_f16(false, a1, false, b1,
                                               (short)0, c, false, false);

    // C/D layout: lane l -> n = l&15 ; vgpr r -> m = r + (l>=16 ? 8 : 0)
    int l = threadIdx.x & 31;
    int n = col0 + (l & 15);
    int mbase = row0 + ((l >= 16) ? 8 : 0);
    float bb = bias ? bias[n] : 0.f;
    float* po = out + (size_t)mbase * C + n;
    if (row0 + 16 <= N) {                 // full tile: straight-line stores
#pragma unroll
        for (int r = 0; r < 8; ++r) {
            float v = c[r] + bb;
            if (slope >= 0.f) v = leakyf(v, slope);
            po[r * C] = v;
        }
    } else {                              // tail tile (N%16!=0 only)
#pragma unroll
        for (int r = 0; r < 8; ++r) {
            if (mbase + r >= N) continue;
            float v = c[r] + bb;
            if (slope >= 0.f) v = leakyf(v, slope);
            po[r * C] = v;
        }
    }
}

// ---- small init / reductions ---------------------------------------------
__global__ void zero_small(float* easum, float* fsum, float* fsq) {
    int t = threadIdx.x;
    if (t < 2)  easum[t] = 0.f;
    if (t < 64) { fsum[t] = 0.f; fsq[t] = 0.f; }
}

__global__ void ea_sum(const float* __restrict__ ea, float* easum, int E) {
    int t = blockIdx.x * blockDim.x + threadIdx.x;
    int T = gridDim.x * blockDim.x;
    float s0 = 0.f, s1 = 0.f;
    for (int e = t; e < E; e += T) { s0 += ea[2 * e]; s1 += ea[2 * e + 1]; }
    atomicAdd(&easum[0], s0);
    atomicAdd(&easum[1], s1);
}

// ce pack per layer (6 floats): [ce0_b0, ce1_b0, ce0_b1, ce1_b1, self_b0, self_b1]
// branch0 = lit (mask==0), branch1 = gcn (mask==1). ce = We @ a_e (2-vector).
__global__ void compute_ce(const float* Wel1, const float* ael1,
                           const float* Weg1, const float* aeg1,
                           const float* Wel2, const float* ael2,
                           const float* Weg2, const float* aeg2,
                           const float* easum, float invE, float* ce) {
    float m0 = easum[0] * invE, m1 = easum[1] * invE;
    const float* Ws[4] = {Wel1, Weg1, Wel2, Weg2};
    const float* As[4] = {ael1, aeg1, ael2, aeg2};
#pragma unroll
    for (int i = 0; i < 4; ++i) {
        float c0 = 0.f, c1 = 0.f;
        for (int f = 0; f < 64; ++f) {
            c0 += Ws[i][f] * As[i][f];
            c1 += Ws[i][64 + f] * As[i][f];
        }
        int layer = i >> 1, br = i & 1;
        float* p = ce + 6 * layer;
        p[2 * br] = c0; p[2 * br + 1] = c1;
        p[4 + br] = m0 * c0 + m1 * c1;
    }
}

// ---- layer-1 node prep: h = x@W for both branches, alphas, init m/denom/ACC
__global__ void node_prep_l1(const float* __restrict__ x,
                             const float* Wl, const float* asl, const float* adl,
                             const float* Wg, const float* asg, const float* adg,
                             float* __restrict__ A0, float* __restrict__ A1,
                             float* as0, float* ad0, float* as1, float* ad1,
                             float* mbuf, float* denom, float* ACC, int N) {
    int wid  = (blockIdx.x * blockDim.x + threadIdx.x) >> 5;
    int lane = threadIdx.x & 31;
    if (wid >= N) return;
    float x0 = x[2 * wid], x1 = x[2 * wid + 1];
    float p0 = 0.f, q0 = 0.f, p1 = 0.f, q1 = 0.f;
#pragma unroll
    for (int j = 0; j < 2; ++j) {
        int f = lane + 32 * j;
        float hl = x0 * Wl[f] + x1 * Wl[64 + f];
        float hg = x0 * Wg[f] + x1 * Wg[64 + f];
        A0[(size_t)wid * 64 + f] = hl;
        A1[(size_t)wid * 64 + f] = hg;
        ACC[(size_t)wid * 64 + f] = 0.f;
        p0 += hl * asl[f]; q0 += hl * adl[f];
        p1 += hg * asg[f]; q1 += hg * adg[f];
    }
#pragma unroll
    for (int off = 16; off; off >>= 1) {
        p0 += __shfl_xor(p0, off); q0 += __shfl_xor(q0, off);
        p1 += __shfl_xor(p1, off); q1 += __shfl_xor(q1, off);
    }
    if (lane == 0) {
        as0[wid] = p0; ad0[wid] = q0; as1[wid] = p1; ad1[wid] = q1;
        mbuf[wid] = -__builtin_inff();
        denom[wid] = 0.f;
    }
}

// ---- layer-2 alphas from precomputed A0/A1 rows + re-init ----------------
__global__ void node_alphas(const float* __restrict__ A0, const float* __restrict__ A1,
                            const float* asl, const float* adl,
                            const float* asg, const float* adg,
                            float* as0, float* ad0, float* as1, float* ad1,
                            float* mbuf, float* denom, float* ACC, int N) {
    int wid  = (blockIdx.x * blockDim.x + threadIdx.x) >> 5;
    int lane = threadIdx.x & 31;
    if (wid >= N) return;
    float p0 = 0.f, q0 = 0.f, p1 = 0.f, q1 = 0.f;
#pragma unroll
    for (int j = 0; j < 2; ++j) {
        int f = lane + 32 * j;
        float hl = A0[(size_t)wid * 64 + f];
        float hg = A1[(size_t)wid * 64 + f];
        ACC[(size_t)wid * 64 + f] = 0.f;
        p0 += hl * asl[f]; q0 += hl * adl[f];
        p1 += hg * asg[f]; q1 += hg * adg[f];
    }
#pragma unroll
    for (int off = 16; off; off >>= 1) {
        p0 += __shfl_xor(p0, off); q0 += __shfl_xor(q0, off);
        p1 += __shfl_xor(p1, off); q1 += __shfl_xor(q1, off);
    }
    if (lane == 0) {
        as0[wid] = p0; ad0[wid] = q0; as1[wid] = p1; ad1[wid] = q1;
        mbuf[wid] = -__builtin_inff();
        denom[wid] = 0.f;
    }
}

// ---- edge pass 1: logits + segment max -----------------------------------
__global__ void edge_logits(const int* __restrict__ ei, const float* __restrict__ eattr,
                            const int* __restrict__ mask,
                            const float* as0, const float* ad0,
                            const float* as1, const float* ad1,
                            const float* __restrict__ ce,
                            float* __restrict__ logits, float* mbuf, int E, int N) {
    int e = blockIdx.x * blockDim.x + threadIdx.x;
    if (e >= E + N) return;
    int s, d; float et;
    if (e < E) {
        s = ei[e]; d = ei[E + e];
        int br = mask[d];
        et = eattr[2 * e] * ce[2 * br] + eattr[2 * e + 1] * ce[2 * br + 1];
        float lg = (br ? (as1[s] + ad1[d]) : (as0[s] + ad0[d])) + et;
        lg = leakyf(lg, NEG_GAT);
        logits[e] = lg;
        atomicMaxFloat(&mbuf[d], lg);
    } else {
        s = d = e - E;                       // self loop, ea = mean(edge_attr)
        int br = mask[d];
        et = ce[4 + br];
        float lg = (br ? (as1[s] + ad1[d]) : (as0[s] + ad0[d])) + et;
        lg = leakyf(lg, NEG_GAT);
        logits[e] = lg;
        atomicMaxFloat(&mbuf[d], lg);
    }
}

// ---- edge pass 2: exp + segment sum --------------------------------------
__global__ void edge_exp(const int* __restrict__ ei, const float* __restrict__ mbuf,
                         float* __restrict__ logits, float* denom, int E, int N) {
    int e = blockIdx.x * blockDim.x + threadIdx.x;
    if (e >= E + N) return;
    int d = (e < E) ? ei[E + e] : (e - E);
    float ex = __expf(logits[e] - mbuf[d]);
    logits[e] = ex;
    atomicAdd(&denom[d], ex);
}

// ---- edge pass 3: weighted aggregation (wave per edge, 64 feats) ---------
__global__ void edge_aggregate(const int* __restrict__ ei, const int* __restrict__ mask,
                               const float* __restrict__ logits, const float* __restrict__ denom,
                               const float* __restrict__ A0, const float* __restrict__ A1,
                               float* __restrict__ ACC, int E, int N) {
    int wid  = (blockIdx.x * blockDim.x + threadIdx.x) >> 5;
    int lane = threadIdx.x & 31;
    if (wid >= E + N) return;
    int s = (wid < E) ? ei[wid]     : (wid - E);
    int d = (wid < E) ? ei[E + wid] : (wid - E);
    float coef = logits[wid] / denom[d];
    const float* H = mask[d] ? A1 : A0;
    atomicAdd(&ACC[(size_t)d * 64 + lane],      H[(size_t)s * 64 + lane]      * coef);
    atomicAdd(&ACC[(size_t)d * 64 + lane + 32], H[(size_t)s * 64 + lane + 32] * coef);
}

// ---- BN stats: add branch bias in place, accumulate per-feature sums -----
__global__ void bn_stats(float* __restrict__ ACC, const int* __restrict__ mask,
                         const float* b0, const float* b1,
                         float* fsum, float* fsq, int N) {
    int t = blockIdx.x * blockDim.x + threadIdx.x;
    int f = t & 63;
    int stride = (gridDim.x * blockDim.x) >> 6;
    float s = 0.f, q = 0.f;
    for (int r = t >> 6; r < N; r += stride) {
        float bias = mask[r] ? b1[f] : b0[f];
        float v = ACC[(size_t)r * 64 + f] + bias;
        ACC[(size_t)r * 64 + f] = v;
        s += v; q += v * v;
    }
    atomicAdd(&fsum[f], s);
    atomicAdd(&fsq[f], q);
}

__global__ void bn_apply(const float* __restrict__ ACC,
                         const float* fsum, const float* fsq,
                         const float* g, const float* beta,
                         float* __restrict__ H, int N) {
    int idx4 = blockIdx.x * blockDim.x + threadIdx.x;   // float4 granularity
    if (idx4 >= N * 16) return;
    int f0 = (idx4 * 4) & 63;
    float invN = 1.f / (float)N;
    float4 v = ((const float4*)ACC)[idx4];
    float* vp = &v.x;
    float4 o;
    float* op = &o.x;
#pragma unroll
    for (int j = 0; j < 4; ++j) {
        int f = f0 + j;
        float mu  = fsum[f] * invN;
        float var = fsq[f] * invN - mu * mu;
        float t = (vp[j] - mu) * rsqrtf(var + BN_EPS) * g[f] + beta[f];
        op[j] = leakyf(t, NEG_ACT);
    }
    ((float4*)H)[idx4] = o;
}

// ---- post layer-2: h = leaky(ACC + b_branch) -----------------------------
__global__ void bias_act(const float* __restrict__ ACC, const int* __restrict__ mask,
                         const float* b0, const float* b1,
                         float* __restrict__ H, int N) {
    int idx4 = blockIdx.x * blockDim.x + threadIdx.x;   // float4 granularity
    if (idx4 >= N * 16) return;
    int r  = idx4 >> 4;
    int f0 = (idx4 * 4) & 63;
    const float* bb = mask[r] ? b1 : b0;
    float4 v = ((const float4*)ACC)[idx4];
    float* vp = &v.x;
    float4 o;
    float* op = &o.x;
#pragma unroll
    for (int j = 0; j < 4; ++j)
        op[j] = leakyf(vp[j] + bb[f0 + j], NEG_ACT);
    ((float4*)H)[idx4] = o;
}

// ---- final: out = (hid @ f2_W + f2_b) * mask (wave per node) -------------
__global__ void mlp_out(const float* __restrict__ HID, const float* __restrict__ w2,
                        const float* b2, const int* __restrict__ mask,
                        float* __restrict__ out, int N) {
    int wid  = (blockIdx.x * blockDim.x + threadIdx.x) >> 5;
    int lane = threadIdx.x & 31;
    if (wid >= N) return;
    float v = HID[(size_t)wid * 32 + lane] * w2[lane];
#pragma unroll
    for (int off = 16; off; off >>= 1) v += __shfl_xor(v, off);
    if (lane == 0) out[wid] = (v + b2[0]) * (float)mask[wid];
}

// ---------------------------------------------------------------------------
extern "C" void kernel_launch(void* const* d_in, const int* in_sizes, int n_in,
                              void* d_out, int out_size, void* d_ws, size_t ws_size,
                              hipStream_t stream) {
    const float* x     = (const float*)d_in[0];
    const float* eattr = (const float*)d_in[1];
    const int*   ei    = (const int*)d_in[2];
    const int*   mask  = (const int*)d_in[3];

    // per-branch param blocks: W, We, as, ad, ae, b
    const float* gcn_W  = (const float*)d_in[4];
    const float* gcn_We = (const float*)d_in[5];
    const float* gcn_as = (const float*)d_in[6];
    const float* gcn_ad = (const float*)d_in[7];
    const float* gcn_ae = (const float*)d_in[8];
    const float* gcn_b  = (const float*)d_in[9];
    const float* lit_W  = (const float*)d_in[10];
    const float* lit_We = (const float*)d_in[11];
    const float* lit_as = (const float*)d_in[12];
    const float* lit_ad = (const float*)d_in[13];
    const float* lit_ae = (const float*)d_in[14];
    const float* lit_b  = (const float*)d_in[15];
    const float* g2_W   = (const float*)d_in[16];
    const float* g2_We  = (const float*)d_in[17];
    const float* g2_as  = (const float*)d_in[18];
    const float* g2_ad  = (const float*)d_in[19];
    const float* g2_ae  = (const float*)d_in[20];
    const float* g2_b   = (const float*)d_in[21];
    const float* l2_W   = (const float*)d_in[22];
    const float* l2_We  = (const float*)d_in[23];
    const float* l2_as  = (const float*)d_in[24];
    const float* l2_ad  = (const float*)d_in[25];
    const float* l2_ae  = (const float*)d_in[26];
    const float* l2_b   = (const float*)d_in[27];
    const float* ln_g   = (const float*)d_in[28];
    const float* ln_b   = (const float*)d_in[29];
    const float* f1_W   = (const float*)d_in[30];
    const float* f1_b   = (const float*)d_in[31];
    const float* f2_W   = (const float*)d_in[32];
    const float* f2_b   = (const float*)d_in[33];

    const int N  = in_sizes[3];
    const int E  = in_sizes[1] / 2;
    const int ET = E + N;

    // ---- workspace layout (floats) ----
    float* ws = (float*)d_ws;
    size_t off = 0;
    float* A0     = ws + off; off += (size_t)N * 64;  // branch0 (lit) features
    float* A1     = ws + off; off += (size_t)N * 64;  // branch1 (gcn) features
    float* ACC    = ws + off; off += (size_t)N * 64;  // aggregation accumulator
    float* HBUF   = ws + off; off += (size_t)N * 64;  // activated node features
    float* HID    = ws + off; off += (size_t)N * 32;  // MLP hidden
    float* logits = ws + off; off += (size_t)ET;
    float* mbuf   = ws + off; off += (size_t)N;
    float* denom  = ws + off; off += (size_t)N;
    float* as0    = ws + off; off += (size_t)N;
    float* ad0    = ws + off; off += (size_t)N;
    float* as1    = ws + off; off += (size_t)N;
    float* ad1    = ws + off; off += (size_t)N;
    float* fsum   = ws + off; off += 64;
    float* fsq    = ws + off; off += 64;
    float* easum  = ws + off; off += 2;
    float* ce     = ws + off; off += 12;              // 6 per layer

    const int ntiles = (N + 15) / 16;
    const int nodeWaves  = (N + 7) / 8;               // 256-thread blocks = 8 waves
    const int edgeWaves  = (ET + 7) / 8;
    const int edgeBlocks = (ET + 255) / 256;
    const int vec4Blocks = (N * 16 + 255) / 256;      // float4-granular kernels

    // ---- shared precompute ----
    zero_small<<<1, 256, 0, stream>>>(easum, fsum, fsq);
    ea_sum<<<256, 256, 0, stream>>>(eattr, easum, E);
    compute_ce<<<1, 1, 0, stream>>>(lit_We, lit_ae, gcn_We, gcn_ae,
                                    l2_We, l2_ae, g2_We, g2_ae,
                                    easum, 1.f / (float)E, ce);

    // ---- layer 1 ----
    node_prep_l1<<<nodeWaves, 256, 0, stream>>>(x,
        lit_W, lit_as, lit_ad, gcn_W, gcn_as, gcn_ad,
        A0, A1, as0, ad0, as1, ad1, mbuf, denom, ACC, N);
    edge_logits<<<edgeBlocks, 256, 0, stream>>>(ei, eattr, mask,
        as0, ad0, as1, ad1, ce + 0, logits, mbuf, E, N);
    edge_exp<<<edgeBlocks, 256, 0, stream>>>(ei, mbuf, logits, denom, E, N);
    edge_aggregate<<<edgeWaves, 256, 0, stream>>>(ei, mask, logits, denom,
        A0, A1, ACC, E, N);
    bn_stats<<<512, 256, 0, stream>>>(ACC, mask, lit_b, gcn_b, fsum, fsq, N);
    bn_apply<<<vec4Blocks, 256, 0, stream>>>(ACC, fsum, fsq, ln_g, ln_b, HBUF, N);

    // ---- layer 2: WMMA GEMMs h@W for both branches ----
    gemm64_wmma<64><<<dim3(ntiles, 4), 32, 0, stream>>>(HBUF, l2_W, nullptr, A0, N, -1.f);
    gemm64_wmma<64><<<dim3(ntiles, 4), 32, 0, stream>>>(HBUF, g2_W, nullptr, A1, N, -1.f);
    node_alphas<<<nodeWaves, 256, 0, stream>>>(A0, A1,
        l2_as, l2_ad, g2_as, g2_ad,
        as0, ad0, as1, ad1, mbuf, denom, ACC, N);
    edge_logits<<<edgeBlocks, 256, 0, stream>>>(ei, eattr, mask,
        as0, ad0, as1, ad1, ce + 6, logits, mbuf, E, N);
    edge_exp<<<edgeBlocks, 256, 0, stream>>>(ei, mbuf, logits, denom, E, N);
    edge_aggregate<<<edgeWaves, 256, 0, stream>>>(ei, mask, logits, denom,
        A0, A1, ACC, E, N);
    bias_act<<<vec4Blocks, 256, 0, stream>>>(ACC, mask, l2_b, g2_b, HBUF, N);

    // ---- MLP head: WMMA GEMM 64->32 + fused leaky, then 32->1 ----
    gemm64_wmma<32><<<dim3(ntiles, 2), 32, 0, stream>>>(HBUF, f1_W, f1_b, HID, N, NEG_ACT);
    mlp_out<<<nodeWaves, 256, 0, stream>>>(HID, f2_W, f2_b, mask, (float*)d_out, N);
}
